// GCNII_91216515432813
// MI455X (gfx1250) — matrix operand order repaired and checked by reading, not compiled, not run on hardware
//
#include <hip/hip_runtime.h>
#include <math.h>

typedef __attribute__((ext_vector_type(16))) __bf16 v16bf;
typedef __attribute__((ext_vector_type(8)))  float  v8f;

#define HID   64
#define FIN   128
#define CLS   47
#define NLAY  64

// ---------------- packed-fragment index helpers (ISA 7.12.2 16-bit layouts) ----
// A (16xK tile, bf16): frag kb covers K=[kb*32, kb*32+32)
//   lane = half*16 + m, elem = ((kr>>4)<<3) | (kr&7), half = (kr>>3)&1, kr = k&31
__device__ __forceinline__ int a_pack_idx(int m, int k) {
    int kb = k >> 5, kr = k & 31;
    int lane = ((kr >> 3) & 1) * 16 + m;
    int elem = ((kr >> 4) << 3) | (kr & 7);
    return (kb * 32 + lane) * 16 + elem;
}
// B (Kx64 weight, bf16): frag (kb,nt); lane = ((kr>>3)&1)*16 + (c&15), elem = ((kr>>4)<<3) | (kr&7)
__device__ __forceinline__ int b_pack_idx(int k, int c) {
    int kb = k >> 5, kr = k & 31;
    int lane = ((kr >> 3) & 1) * 16 + (c & 15);
    int nt = c >> 4;
    return ((kb * 4 + nt) * 32 + lane) * 16 + (((kr >> 4) << 3) | (kr & 7));
}

// ---------------- CSR build ----------------
__global__ void k_zero_int(int* a, int n) {
    int i = blockIdx.x * 256 + threadIdx.x;
    if (i < n) a[i] = 0;
}
__global__ void k_count(const int* __restrict__ ei, int* cnt, int E) {
    int e = blockIdx.x * 256 + threadIdx.x;
    if (e < E) atomicAdd(&cnt[ei[(size_t)E + e]], 1);
}
__global__ void k_dinv(const int* __restrict__ cnt, float* __restrict__ dinv, int n) {
    int i = blockIdx.x * 256 + threadIdx.x;
    if (i < n) dinv[i] = rsqrtf(1.0f + (float)cnt[i]);   // +1: self loop
}
// hierarchical exclusive scan: 1024 elems/block
__global__ __launch_bounds__(256) void k_scan_local(const int* __restrict__ cnt,
        int* __restrict__ rowptr, int* __restrict__ bsum, int n) {
    __shared__ int tmp[256];
    int t = threadIdx.x;
    int i0 = blockIdx.x * 1024 + t * 4;
    int a0 = (i0 + 0 < n) ? cnt[i0 + 0] : 0;
    int a1 = (i0 + 1 < n) ? cnt[i0 + 1] : 0;
    int a2 = (i0 + 2 < n) ? cnt[i0 + 2] : 0;
    int a3 = (i0 + 3 < n) ? cnt[i0 + 3] : 0;
    int s4 = a0 + a1 + a2 + a3;
    tmp[t] = s4;
    __syncthreads();
    for (int d = 1; d < 256; d <<= 1) {
        int v = (t >= d) ? tmp[t - d] : 0;
        __syncthreads();
        tmp[t] += v;
        __syncthreads();
    }
    int excl = tmp[t] - s4;
    if (i0 + 0 < n) rowptr[i0 + 0] = excl;
    if (i0 + 1 < n) rowptr[i0 + 1] = excl + a0;
    if (i0 + 2 < n) rowptr[i0 + 2] = excl + a0 + a1;
    if (i0 + 3 < n) rowptr[i0 + 3] = excl + a0 + a1 + a2;
    if (t == 255) bsum[blockIdx.x] = tmp[255];
}
__global__ void k_scan_blocks(int* bs, int nb, int* rowptr, int n, int E) {
    int run = 0;
    for (int i = 0; i < nb; ++i) { int v = bs[i]; bs[i] = run; run += v; }
    rowptr[n] = E;
}
__global__ void k_scan_add(int* rowptr, const int* __restrict__ bs, int n) {
    int i = blockIdx.x * 256 + threadIdx.x;
    if (i < n) rowptr[i] += bs[blockIdx.x >> 2];     // 4 blocks of 256 per 1024-chunk
}
__global__ void k_copy_int(const int* __restrict__ a, int* __restrict__ b, int n) {
    int i = blockIdx.x * 256 + threadIdx.x;
    if (i < n) b[i] = a[i];
}
__global__ void k_scatter(const int* __restrict__ ei, int* cursor,
                          int* __restrict__ csrc, int E) {
    int e = blockIdx.x * 256 + threadIdx.x;
    if (e >= E) return;
    int dst = ei[(size_t)E + e];
    int pos = atomicAdd(&cursor[dst], 1);
    csrc[pos] = ei[e];
}

// ---------------- input GEMM: x0 = relu(x @ W1 + b1) ----------------
__global__ __launch_bounds__(256) void k_in_gemm(const float* __restrict__ x,
        const float* __restrict__ W1, const float* __restrict__ b1,
        float* __restrict__ x0, int n) {
    __shared__ v16bf sW[16 * 32];       // 16 frags (kb0..3 x nt0..3), 16 KB
    __shared__ v16bf sA[8][4 * 32];     // per-wave: 4 kb frags, 32 KB total
    int tid = threadIdx.x;
    __bf16* wp = (__bf16*)sW;
    for (int i = tid; i < FIN * HID; i += 256)
        wp[b_pack_idx(i >> 6, i & 63)] = (__bf16)W1[i];
    __syncthreads();
    int wave = tid >> 5, lane = tid & 31;
    int row0 = (blockIdx.x * 8 + wave) * 16;
    if (row0 >= n) return;
    __bf16* ap = (__bf16*)sA[wave];
    for (int i = lane; i < 16 * 32; i += 32) {        // 16 rows x 128 cols / 4
        int m = i >> 5, c4 = (i & 31) * 4;
        float4 vv = *(const float4*)(x + (size_t)(row0 + m) * FIN + c4);
        float f[4] = {vv.x, vv.y, vv.z, vv.w};
#pragma unroll
        for (int j = 0; j < 4; ++j) ap[a_pack_idx(m, c4 + j)] = (__bf16)f[j];
    }
    int m = lane & 15, half = lane >> 4;
    for (int nt = 0; nt < 4; ++nt) {
        v8f acc = {};
#pragma unroll
        for (int kb = 0; kb < 4; ++kb) {
            v16bf a = sA[wave][kb * 32 + lane];
            v16bf b = sW[(kb * 4 + nt) * 32 + lane];
            acc = __builtin_amdgcn_wmma_f32_16x16x32_bf16(false, a, false, b,
                                                          (short)0, acc, false, false);
        }
#pragma unroll
        for (int r = 0; r < 8; ++r) {
            int rr = r + (half ? 8 : 0);
            int col = nt * 16 + m;
            float val = acc[r] + b1[col];
            x0[(size_t)(row0 + rr) * HID + col] = val > 0.f ? val : 0.f;
        }
    }
}

// ---------------- SpMM gather: p = D^-1/2 (A+I) D^-1/2 @ h (no atomics) ------
__global__ __launch_bounds__(256) void k_spmm_csr(const int* __restrict__ rowptr,
        const int* __restrict__ csrc, const float* __restrict__ dinv,
        const float* __restrict__ h, float* __restrict__ p, int n) {
    int wave = threadIdx.x >> 5, lane = threadIdx.x & 31;
    int row = blockIdx.x * 8 + wave;
    if (row >= n) return;
    float dr = dinv[row];
    float2 acc = ((const float2*)(h + (size_t)row * HID))[lane];
    acc.x *= dr * dr; acc.y *= dr * dr;                 // self-loop term
    int beg = rowptr[row], end = rowptr[row + 1];
    __builtin_prefetch(csrc + beg, 0, 1);               // global_prefetch_b8
    for (int j = beg; j < end; ++j) {
        int src = csrc[j];
        float we = dr * dinv[src];
        float2 hv = ((const float2*)(h + (size_t)src * HID))[lane];
        acc.x += we * hv.x;
        acc.y += we * hv.y;
    }
    ((float2*)(p + (size_t)row * HID))[lane] = acc;
}

// -------- fused layer: s = 0.9p + 0.1x0 ; h = relu((1-b)s + b*(s@W)) --------
__global__ __launch_bounds__(256) void k_layer(const float* __restrict__ p,
        const float* __restrict__ x0, const float* __restrict__ Wl,
        float* __restrict__ hout, float beta, int n) {
    __shared__ v16bf sW[8 * 32];        // 8 frags (kb0..1 x nt0..3), 8 KB
    __shared__ v16bf sA[8][2 * 32];     // per-wave packed s (bf16), 16 KB
    __shared__ float sS[8][16 * HID];   // per-wave s (f32, bypass), 32 KB
    int tid = threadIdx.x;
    __bf16* wp = (__bf16*)sW;
    for (int i = tid; i < HID * HID; i += 256)
        wp[b_pack_idx(i >> 6, i & 63)] = (__bf16)Wl[i];
    __syncthreads();
    int wave = tid >> 5, lane = tid & 31;
    int row0 = (blockIdx.x * 8 + wave) * 16;
    if (row0 >= n) return;
    __bf16* ap = (__bf16*)sA[wave];
    float*  ss = sS[wave];
    for (int i = lane; i < 16 * 16; i += 32) {          // 16 rows x 64 cols / 4
        int m = i >> 4, c4 = (i & 15) * 4;
        size_t off = (size_t)(row0 + m) * HID + c4;
        float4 pv = *(const float4*)(p + off);
        float4 xv = *(const float4*)(x0 + off);
        float4 sv;
        sv.x = 0.9f * pv.x + 0.1f * xv.x;
        sv.y = 0.9f * pv.y + 0.1f * xv.y;
        sv.z = 0.9f * pv.z + 0.1f * xv.z;
        sv.w = 0.9f * pv.w + 0.1f * xv.w;
        *(float4*)&ss[m * HID + c4] = sv;
        float f[4] = {sv.x, sv.y, sv.z, sv.w};
#pragma unroll
        for (int j = 0; j < 4; ++j) ap[a_pack_idx(m, c4 + j)] = (__bf16)f[j];
    }
    int m = lane & 15, half = lane >> 4;
    float omb = 1.0f - beta;
    for (int nt = 0; nt < 4; ++nt) {
        v8f acc = {};
#pragma unroll
        for (int kb = 0; kb < 2; ++kb) {
            v16bf a = sA[wave][kb * 32 + lane];
            v16bf b = sW[(kb * 4 + nt) * 32 + lane];
            acc = __builtin_amdgcn_wmma_f32_16x16x32_bf16(false, a, false, b,
                                                          (short)0, acc, false, false);
        }
#pragma unroll
        for (int r = 0; r < 8; ++r) {
            int rr = r + (half ? 8 : 0);
            int col = nt * 16 + m;
            float val = omb * ss[rr * HID + col] + beta * acc[r];
            hout[(size_t)(row0 + rr) * HID + col] = val > 0.f ? val : 0.f;
        }
    }
}

// ---------------- output: log_softmax(h @ W2 + b2) ----------------
__global__ __launch_bounds__(256) void k_out(const float* __restrict__ h,
        const float* __restrict__ W2, const float* __restrict__ b2,
        float* __restrict__ out, int n) {
    __shared__ float sW[HID * CLS + CLS];
    __shared__ float sR[8][HID];
    int tid = threadIdx.x;
    for (int i = tid; i < HID * CLS; i += 256) sW[i] = W2[i];
    if (tid < CLS) sW[HID * CLS + tid] = b2[tid];
    __syncthreads();
    int wave = tid >> 5, lane = tid & 31;
    int row = blockIdx.x * 8 + wave;
    if (row >= n) return;
    float* hr = sR[wave];
    if (lane < 16) {
        float4 v = *(const float4*)(h + (size_t)row * HID + lane * 4);
        hr[lane * 4 + 0] = v.x; hr[lane * 4 + 1] = v.y;
        hr[lane * 4 + 2] = v.z; hr[lane * 4 + 3] = v.w;
    }
    float logit = -3.4e38f;
    if (lane < CLS) {
        float acc = sW[HID * CLS + lane];
#pragma unroll 8
        for (int k = 0; k < HID; ++k) acc += hr[k] * sW[k * CLS + lane];
        logit = acc;
    }
    float mx = logit;
    for (int off = 16; off; off >>= 1) mx = fmaxf(mx, __shfl_xor(mx, off, 32));
    float ex = (lane < CLS) ? __expf(logit - mx) : 0.f;
    float sum = ex;
    for (int off = 16; off; off >>= 1) sum += __shfl_xor(sum, off, 32);
    if (lane < CLS) out[(size_t)row * CLS + lane] = (logit - mx) - __logf(sum);
}

// ---------------- host orchestration ----------------
static inline size_t alignup(size_t v) { return (v + 255) & ~(size_t)255; }

extern "C" void kernel_launch(void* const* d_in, const int* in_sizes, int n_in,
                              void* d_out, int out_size, void* d_ws, size_t ws_size,
                              hipStream_t stream) {
    const float* x  = (const float*)d_in[0];
    const int*   ei = (const int*)d_in[1];
    const float* W1 = (const float*)d_in[2];
    const float* b1 = (const float*)d_in[3];
    const float* cW = (const float*)d_in[4];
    const float* W2 = (const float*)d_in[5];
    const float* b2 = (const float*)d_in[6];
    const int N = in_sizes[0] / FIN;
    const int E = in_sizes[1] / 2;
    const int nchunks = (N + 1023) / 1024;

    char* ws = (char*)d_ws;
    float* dinv   = (float*)ws; ws += alignup((size_t)N * 4);
    int*   rowptr = (int*)ws;   ws += alignup((size_t)(N + 1) * 4);
    int*   cursor = (int*)ws;   ws += alignup((size_t)N * 4);   // also the count buf
    int*   bsum   = (int*)ws;   ws += alignup((size_t)nchunks * 4);
    int*   csrc   = (int*)ws;   ws += alignup((size_t)E * 4);
    float* x0     = (float*)ws; ws += (size_t)N * HID * 4;
    float* hA     = (float*)ws; ws += (size_t)N * HID * 4;
    float* hB     = (float*)ws; ws += (size_t)N * HID * 4;
    float* p      = (float*)ws;

    const int bN   = (N + 255) / 256;
    const int bE   = (E + 255) / 256;
    const int bGem = ((N + 15) / 16 + 7) / 8;
    const int bRow = (N + 7) / 8;

    // CSR build (cursor doubles as the count buffer)
    k_zero_int   <<<bN, 256, 0, stream>>>(cursor, N);
    k_count      <<<bE, 256, 0, stream>>>(ei, cursor, E);
    k_dinv       <<<bN, 256, 0, stream>>>(cursor, dinv, N);
    k_scan_local <<<nchunks, 256, 0, stream>>>(cursor, rowptr, bsum, N);
    k_scan_blocks<<<1, 1, 0, stream>>>(bsum, nchunks, rowptr, N, E);
    k_scan_add   <<<nchunks * 4, 256, 0, stream>>>(rowptr, bsum, N);
    k_copy_int   <<<bN, 256, 0, stream>>>(rowptr, cursor, N);
    k_scatter    <<<bE, 256, 0, stream>>>(ei, cursor, csrc, E);

    k_in_gemm<<<bGem, 256, 0, stream>>>(x, W1, b1, x0, N);

    const float* hin = x0;
    float* hout = hA;
    for (int l = 0; l < NLAY; ++l) {
        k_spmm_csr<<<bRow, 256, 0, stream>>>(rowptr, csrc, dinv, hin, p, N);
        float beta = logf(0.5f / (float)(l + 1) + 1.0f);
        k_layer<<<bGem, 256, 0, stream>>>(p, x0, cW + (size_t)l * HID * HID, hout, beta, N);
        hin = hout;
        hout = (hout == hA) ? hB : hA;
    }
    k_out<<<bRow, 256, 0, stream>>>(hin, W2, b2, (float*)d_out, N);
}
